// PixelTransformer_86646670230185
// MI455X (gfx1250) — matrix operand order repaired and checked by reading, not compiled, not run on hardware
//
#include <hip/hip_runtime.h>
#include <hip/hip_bf16.h>

typedef _Float16 half_t;
typedef __attribute__((ext_vector_type(16))) _Float16 v16h;
typedef __attribute__((ext_vector_type(8)))  _Float16 v8h;
typedef __attribute__((ext_vector_type(8)))  float    v8f;
typedef unsigned int __attribute__((ext_vector_type(4))) v4u;
typedef int __attribute__((ext_vector_type(8))) v8i;
typedef int __attribute__((ext_vector_type(4))) v4i;

// ---------------------------------------------------------------------------
// WMMA GEMM: C[M,N] = scale * (A[M,K] @ B[K,N]) + bias, optional ReLU.
// A is f32 [M,K] row-major (activations), staged through VALU f32->f16.
// BT is f16 [N,K] row-major (B transposed); its 64x32-half tile is moved by
// the Tensor Data Mover (tensor_load_to_lds) with D# padding producing the
// 40-half LDS stride directly.  4 v_wmma_f32_16x16x32_f16 per K-step/wave.
// ---------------------------------------------------------------------------
__global__ __launch_bounds__(256) void wmma_gemm_kernel(
    const float* __restrict__ A, const half_t* __restrict__ BT,
    const float* __restrict__ bias, float* __restrict__ C,
    int M, int N, int K, float scale, int relu)
{
    constexpr int BM = 128, BN = 64, BK = 32;
    constexpr int LA = 40;   // As row stride (halves): 80B, keeps 16B alignment
    constexpr int LB = 40;   // Bs column stride (halves): 32 data + 4-DWORD TDM pad
    __shared__ half_t As[BM * LA];   // As[m][k]
    __shared__ half_t Bs[BN * LB];   // Bs[n][k]  (column of B = row of BT)

    const int tid  = threadIdx.x;
    const int lane = tid & 31;
    const int wave = tid >> 5;
    const int hs   = lane >> 4;     // half-wave select per 16-bit WMMA layouts
    const int l15  = lane & 15;

    const int bm = blockIdx.y * BM;
    const int bn = blockIdx.x * BN;
    const int wm0 = (wave & 3) * 16;        // 0,16,32,48
    const int wm1 = wm0 + 64;               // second M tile
    const int wn  = (wave >> 2) * 16;       // 0,16 ; N tiles at wn and wn+32

    v8f acc00 = {}, acc01 = {}, acc10 = {}, acc11 = {};

    // A staging assignment: 128 rows x 32 cols, 16 f32 per thread
    const int ar = tid >> 1, ac = (tid & 1) * 16;
    const unsigned bs_lds = (unsigned)(size_t)(&Bs[0]);   // LDS byte address (low 32 bits)

    for (int k0 = 0; k0 < K; k0 += BK) {
        // --- TDM: one wave issues the async tensor DMA for the whole B tile.
        // Tile = 64 rows x 32 halves from BT[bn..bn+64){K}, row stride K elems.
        // pad_enable inserts 4 DWORDs after every 16 DWORDs -> LDS stride 40 halves.
        if (wave == 0) {
            const unsigned long long ga =
                (unsigned long long)(size_t)(BT + (size_t)bn * K + k0);
            v4u g0 = { 1u,                                   // count=1 (valid user D#)
                       bs_lds,                               // lds_addr
                       (unsigned)(ga & 0xffffffffu),         // global_addr[31:0]
                       (unsigned)((ga >> 32) & 0x01ffffffu)  // global_addr[56:32]
                           | 0x80000000u };                  // type=2 ("image")
            v8i g1 = { (int)((1u << 16)      // data_size = 2 bytes
                           | (1u << 20)      // pad_enable
                           | (3u << 22)      // pad_interval: 16 DWORDs
                           | (3u << 25)),    // pad_amount: 4 DWORDs
                       (int)(32u << 16),     // tensor_dim0 = 32 (bits 79:48 lo)
                       (int)(64u << 16),     // tensor_dim1 = 64 (bits 111:80 lo)
                       (int)(32u << 16),     // tile_dim0 = 32   (bits 127:112)
                       (int)64,              // tile_dim1 = 64   (bits 143:128)
                       (int)K,               // tensor_dim0_stride = K (bits 207:160 lo)
                       0, 0 };
            v4i g2 = { 0, 0, 0, 0 };
            v4i g3 = { 0, 0, 0, 0 };
            v8i g4 = { 0, 0, 0, 0, 0, 0, 0, 0 };
            __builtin_amdgcn_tensor_load_to_lds(g0, g1, g2, g3, g4, 0);
        }

        // --- Stage A (f32 -> f16) into LDS, row-major [m][k]
        const float* ap = A + (size_t)(bm + ar) * K + k0 + ac;
        half_t* asp = &As[ar * LA + ac];
#pragma unroll
        for (int i = 0; i < 16; ++i) asp[i] = (half_t)ap[i];
        if (k0 + BK < K)                       // global_prefetch_b8 for next A tile
            __builtin_prefetch(ap + BK, 0, 1);

        if (wave == 0) __builtin_amdgcn_s_wait_tensorcnt(0);
        __syncthreads();

        // A fragments (16x32 f16 layout: lanes0-15 K{0..7,16..23}, lanes16-31 K{8..15,24..31})
        const int kb = hs ? 8 : 0;
        v16h af0, af1;
        {
            const half_t* arow = &As[(wm0 + l15) * LA];
            v8h lo = *(const v8h*)(arow + kb);
            v8h hi = *(const v8h*)(arow + kb + 16);
            af0 = __builtin_shufflevector(lo, hi, 0,1,2,3,4,5,6,7,8,9,10,11,12,13,14,15);
            arow = &As[(wm1 + l15) * LA];
            lo = *(const v8h*)(arow + kb);
            hi = *(const v8h*)(arow + kb + 16);
            af1 = __builtin_shufflevector(lo, hi, 0,1,2,3,4,5,6,7,8,9,10,11,12,13,14,15);
        }
        // B fragments (32x16: lane holds one column; lanes0-15 K{0..7,16..23}, lanes16-31 K{8..15,24..31})
        const int kbb = hs * 8;
        v16h bf0, bf1;
        {
            const half_t* bcol = &Bs[(wn + l15) * LB + kbb];
            v8h lo = *(const v8h*)bcol;
            v8h hi = *(const v8h*)(bcol + 16);
            bf0 = __builtin_shufflevector(lo, hi, 0,1,2,3,4,5,6,7,8,9,10,11,12,13,14,15);
            bcol = &Bs[(wn + 32 + l15) * LB + kbb];
            lo = *(const v8h*)bcol;
            hi = *(const v8h*)(bcol + 16);
            bf1 = __builtin_shufflevector(lo, hi, 0,1,2,3,4,5,6,7,8,9,10,11,12,13,14,15);
        }
        acc00 = __builtin_amdgcn_wmma_f32_16x16x32_f16(false, af0, false, bf0, (short)0, acc00, false, false);
        acc01 = __builtin_amdgcn_wmma_f32_16x16x32_f16(false, af0, false, bf1, (short)0, acc01, false, false);
        acc10 = __builtin_amdgcn_wmma_f32_16x16x32_f16(false, af1, false, bf0, (short)0, acc10, false, false);
        acc11 = __builtin_amdgcn_wmma_f32_16x16x32_f16(false, af1, false, bf1, (short)0, acc11, false, false);
        __syncthreads();
    }

    // C layout: VGPR r -> row (wm + hs*8 + r), col = lane&15
    const int row0 = bm + wm0 + hs * 8;
    const int row1 = bm + wm1 + hs * 8;
    const int c0   = bn + wn + l15;
    const int c1   = c0 + 32;
    const float b0 = bias ? bias[c0] : 0.f;
    const float b1 = bias ? bias[c1] : 0.f;
#pragma unroll
    for (int r = 0; r < 8; ++r) {
        float v00 = acc00[r] * scale + b0;
        float v01 = acc01[r] * scale + b1;
        float v10 = acc10[r] * scale + b0;
        float v11 = acc11[r] * scale + b1;
        if (relu) {
            v00 = fmaxf(v00, 0.f); v01 = fmaxf(v01, 0.f);
            v10 = fmaxf(v10, 0.f); v11 = fmaxf(v11, 0.f);
        }
        C[(size_t)(row0 + r) * N + c0] = v00;
        C[(size_t)(row0 + r) * N + c1] = v01;
        C[(size_t)(row1 + r) * N + c0] = v10;
        C[(size_t)(row1 + r) * N + c1] = v11;
    }
}

// ---------------------------------------------------------------------------
// Elementwise / small kernels
// ---------------------------------------------------------------------------
__global__ void add_scaled_kernel(const float* __restrict__ a, const float* __restrict__ b,
                                  float s, float* __restrict__ y, size_t n) {
    size_t i = (size_t)blockIdx.x * 256 + threadIdx.x;
    if (i < n) y[i] = a[i] + s * b[i];
}
__global__ void add_inplace_kernel(float* __restrict__ y, const float* __restrict__ x, size_t n) {
    size_t i = (size_t)blockIdx.x * 256 + threadIdx.x;
    if (i < n) y[i] += x[i];
}
__global__ void convert_f16_kernel(const float* __restrict__ in, half_t* __restrict__ out, size_t n) {
    size_t i = (size_t)blockIdx.x * 256 + threadIdx.x;
    if (i < n) out[i] = (half_t)in[i];
}
// in [R,C] f32 -> out [C,R] f16
__global__ void transpose_f16_kernel(const float* __restrict__ in, half_t* __restrict__ out,
                                     int R, int C) {
    size_t i = (size_t)blockIdx.x * 256 + threadIdx.x;
    if (i >= (size_t)R * C) return;
    int r = (int)(i / C), c = (int)(i % C);
    out[(size_t)c * R + r] = (half_t)in[i];
}

// LayerNorm over D=256, one row per 256-thread block (8 wave32)
__global__ __launch_bounds__(256) void layernorm_kernel(
    const float* __restrict__ x, const float* __restrict__ g,
    const float* __restrict__ b, float* __restrict__ y) {
    __shared__ float red[256];
    const int row = blockIdx.x, tid = threadIdx.x;
    float v = x[(size_t)row * 256 + tid];
    red[tid] = v; __syncthreads();
    for (int s = 128; s > 0; s >>= 1) { if (tid < s) red[tid] += red[tid + s]; __syncthreads(); }
    float mean = red[0] * (1.f / 256.f); __syncthreads();
    float d = v - mean;
    red[tid] = d * d; __syncthreads();
    for (int s = 128; s > 0; s >>= 1) { if (tid < s) red[tid] += red[tid + s]; __syncthreads(); }
    float var = red[0] * (1.f / 256.f);
    y[(size_t)row * 256 + tid] = d * rsqrtf(var + 1e-5f) * g[tid] + b[tid];
}

// Axial RoPE: x[L,256]; 128 pairs/row; w=64, freq_i = theta^{-i/64}
__global__ __launch_bounds__(128) void rope_kernel(float* __restrict__ x, int mod) {
    const int row = blockIdx.x, j = threadIdx.x;
    const int t = row % mod;
    const float pos  = (j < 64) ? (float)(t & 63) : (float)(t >> 6);
    const int   fi   = (j < 64) ? j : (j - 64);
    const float freq = __powf(10000.f, -(float)fi * (1.f / 64.f));
    const float ang  = pos * freq;
    float sn, cs; __sincosf(ang, &sn, &cs);
    const size_t base = (size_t)row * 256 + 2 * j;
    const float xr = x[base], xi = x[base + 1];
    x[base]     = xr * cs - xi * sn;
    x[base + 1] = xr * sn + xi * cs;
}

// Exact top-k threshold per row: LDS-resident radix binary search (monotone
// float->uint keys), 32 counting passes over 8192 keys held in LDS.
__global__ __launch_bounds__(256) void topk_thresh_kernel(
    const float* __restrict__ scores, int Lk, const int* __restrict__ topk_ptr,
    float* __restrict__ thresh) {
    __shared__ unsigned keys[8192];
    __shared__ int red[256];
    const int row = blockIdx.x, tid = threadIdx.x;
    const int k = *topk_ptr;
    if (k >= Lk) { if (tid == 0) thresh[row] = -__builtin_inff(); return; }
    const float* srow = scores + (size_t)row * Lk;
    for (int i = tid; i < Lk; i += 256) {
        unsigned u = __float_as_uint(srow[i]);
        keys[i] = (u & 0x80000000u) ? ~u : (u | 0x80000000u);
    }
    __syncthreads();
    unsigned result = 0;
    for (int bit = 31; bit >= 0; --bit) {
        const unsigned cand = result | (1u << bit);
        int cnt = 0;
        for (int i = tid; i < Lk; i += 256) cnt += (keys[i] >= cand);
        red[tid] = cnt; __syncthreads();
        for (int s = 128; s > 0; s >>= 1) { if (tid < s) red[tid] += red[tid + s]; __syncthreads(); }
        const int total = red[0]; __syncthreads();
        if (total >= k) result = cand;
    }
    if (tid == 0) {
        unsigned u = result;
        unsigned fb = (u & 0x80000000u) ? (u ^ 0x80000000u) : ~u;
        thresh[row] = __uint_as_float(fb);
    }
}

// Row softmax with optional >=threshold mask (masked -> 0 prob)
__global__ __launch_bounds__(256) void softmax_kernel(
    float* __restrict__ sc, int len, const float* __restrict__ thresh) {
    __shared__ float red[256];
    const int row = blockIdx.x, tid = threadIdx.x;
    float* s = sc + (size_t)row * len;
    const float th = thresh ? thresh[row] : -__builtin_inff();
    float mx = -__builtin_inff();
    for (int i = tid; i < len; i += 256) { float v = s[i]; if (v >= th) mx = fmaxf(mx, v); }
    red[tid] = mx; __syncthreads();
    for (int st = 128; st > 0; st >>= 1) { if (tid < st) red[tid] = fmaxf(red[tid], red[tid + st]); __syncthreads(); }
    mx = red[0]; __syncthreads();
    float sum = 0.f;
    for (int i = tid; i < len; i += 256) { float v = s[i]; sum += (v >= th) ? __expf(v - mx) : 0.f; }
    red[tid] = sum; __syncthreads();
    for (int st = 128; st > 0; st >>= 1) { if (tid < st) red[tid] += red[tid + st]; __syncthreads(); }
    sum = red[0]; __syncthreads();
    const float inv = 1.f / sum;
    for (int i = tid; i < len; i += 256) {
        float v = s[i];
        s[i] = (v >= th) ? __expf(v - mx) * inv : 0.f;
    }
}

__global__ void percent_kernel(const int* __restrict__ topk, int Lk, float* __restrict__ out) {
    const int k = *topk;
    out[0] = (k < Lk) ? ((float)k / (float)Lk) : 1.0f;
}

// ---------------------------------------------------------------------------
// Host orchestration
// ---------------------------------------------------------------------------
extern "C" void kernel_launch(void* const* d_in, const int* in_sizes, int n_in,
                              void* d_out, int out_size, void* d_ws, size_t ws_size,
                              hipStream_t stream) {
    const int Lq = 4096, Lk = 8192, D = 256, FFD = 2048, KVIN = 64;

    const float* curr       = (const float*)d_in[0];
    const float* memory     = (const float*)d_in[1];
    const float* curr_pos   = (const float*)d_in[2];
    const float* memory_pos = (const float*)d_in[3];
    // params flattened in sorted-key (jax pytree) order: fnb, fng, 4x layer(26), topk
    const int PB = 4;
    const float* fnb = (const float*)d_in[PB + 0];
    const float* fng = (const float*)d_in[PB + 1];
    const int*  topk = (const int*)d_in[PB + 2 + 4 * 26];
    (void)in_sizes; (void)n_in; (void)out_size; (void)ws_size;

    // Workspace carve-out
    char* w = (char*)d_ws;
    auto alloc = [&](size_t bytes) { char* p = w; w += (bytes + 255) & ~(size_t)255; return p; };
    float*  X    = (float*)alloc((size_t)Lq * D * 4);
    float*  T2   = (float*)alloc((size_t)Lq * D * 4);
    float*  Qb   = (float*)alloc((size_t)Lq * D * 4);
    float*  Kb   = (float*)alloc((size_t)Lk * D * 4);
    float*  Vb   = (float*)alloc((size_t)Lk * D * 4);
    float*  AO   = (float*)alloc((size_t)Lq * D * 4);
    float*  PR   = (float*)alloc((size_t)Lq * D * 4);
    float*  H    = (float*)alloc((size_t)Lq * FFD * 4);
    float*  SC   = (float*)alloc((size_t)Lq * Lk * 4);
    float*  TH   = (float*)alloc((size_t)Lq * 4);
    float*  KIN  = (float*)alloc((size_t)Lk * KVIN * 4);
    half_t* WT16 = (half_t*)alloc((size_t)D * FFD * 2);   // transposed weight [N,K]
    half_t* BK16 = (half_t*)alloc((size_t)Lk * D * 2);    // K matrix f16 (Bt for scores)
    half_t* VT16 = (half_t*)alloc((size_t)D * Lk * 2);    // V^T f16 (Bt for attn@V)

    auto blocks = [](size_t n) { return (unsigned)((n + 255) / 256); };
    auto conv = [&](const float* src, half_t* dst, size_t n) {
        convert_f16_kernel<<<blocks(n), 256, 0, stream>>>(src, dst, n);
    };
    // B given as [K,N] f32 -> BT f16 [N,K]
    auto transp = [&](const float* src, half_t* dst, int R, int C) {
        transpose_f16_kernel<<<blocks((size_t)R * C), 256, 0, stream>>>(src, dst, R, C);
    };
    auto gemm = [&](const float* A, const half_t* BT, const float* bias, float* C,
                    int M, int N, int K, float scale, int relu) {
        dim3 g(N / 64, M / 128);
        wmma_gemm_kernel<<<g, 256, 0, stream>>>(A, BT, bias, C, M, N, K, scale, relu);
    };
    auto ln = [&](const float* x, const float* g_, const float* b_, float* y, int rows) {
        layernorm_kernel<<<rows, 256, 0, stream>>>(x, g_, b_, y);
    };

    // tgt = curr + 0.1*curr_pos ; kin = memory + memory_pos (same each layer)
    add_scaled_kernel<<<blocks((size_t)Lq * D), 256, 0, stream>>>(curr, curr_pos, 0.1f, X, (size_t)Lq * D);
    add_scaled_kernel<<<blocks((size_t)Lk * KVIN), 256, 0, stream>>>(memory, memory_pos, 1.0f, KIN, (size_t)Lk * KVIN);

    const float inv_sqrt_hd = 1.0f / 16.0f;   // 1/sqrt(256)

    for (int l = 0; l < 4; ++l) {
        auto P = [&](int i) { return (const float*)d_in[PB + 2 + l * 26 + i]; };
        // layer leaf order: 0 ca.kb 1 ca.kw 2 ca.ob 3 ca.ow 4 ca.qb 5 ca.qw 6 ca.vb 7 ca.vw
        //                   8 l1b 9 l1w 10 l2b 11 l2w 12 n1b 13 n1g 14 n2b 15 n2g 16 n3b 17 n3g
        //                   18 sa.kb 19 sa.kw 20 sa.ob 21 sa.ow 22 sa.qb 23 sa.qw 24 sa.vb 25 sa.vw

        // ---- Self-attention ----
        ln(X, P(13), P(12), T2, Lq);
        transp(P(23), WT16, D, D); gemm(T2, WT16, P(22), Qb, Lq, D, D, 1.f, 0);
        transp(P(19), WT16, D, D); gemm(T2, WT16, P(18), Kb, Lq, D, D, 1.f, 0);
        transp(P(25), WT16, D, D); gemm(T2, WT16, P(24), Vb, Lq, D, D, 1.f, 0);
        rope_kernel<<<Lq, 128, 0, stream>>>(Qb, 4096);
        rope_kernel<<<Lq, 128, 0, stream>>>(Kb, 4096);
        conv(Kb, BK16, (size_t)Lq * D);          // Bt of K^T is K itself
        transp(Vb, VT16, Lq, D);                 // Bt of V is V^T
        gemm(Qb, BK16, nullptr, SC, Lq, Lq, D, inv_sqrt_hd, 0);
        softmax_kernel<<<Lq, 256, 0, stream>>>(SC, Lq, nullptr);
        gemm(SC, VT16, nullptr, AO, Lq, D, Lq, 1.f, 0);
        transp(P(21), WT16, D, D); gemm(AO, WT16, P(20), PR, Lq, D, D, 1.f, 0);
        add_inplace_kernel<<<blocks((size_t)Lq * D), 256, 0, stream>>>(X, PR, (size_t)Lq * D);

        // ---- Cross-attention (rope_k_repeat, top-k) ----
        ln(X, P(15), P(14), T2, Lq);
        transp(P(5), WT16, D, D);    gemm(T2,     WT16, P(4), Qb, Lq, D, D,    1.f, 0);
        transp(P(1), WT16, KVIN, D); gemm(KIN,    WT16, P(0), Kb, Lk, D, KVIN, 1.f, 0);
        transp(P(7), WT16, KVIN, D); gemm(memory, WT16, P(6), Vb, Lk, D, KVIN, 1.f, 0);
        rope_kernel<<<Lq, 128, 0, stream>>>(Qb, 4096);
        rope_kernel<<<Lk, 128, 0, stream>>>(Kb, 4096);   // tile(cos,(2,1)) == t % 4096
        conv(Kb, BK16, (size_t)Lk * D);
        transp(Vb, VT16, Lk, D);
        gemm(Qb, BK16, nullptr, SC, Lq, Lk, D, inv_sqrt_hd, 0);
        topk_thresh_kernel<<<Lq, 256, 0, stream>>>(SC, Lk, topk, TH);
        softmax_kernel<<<Lq, 256, 0, stream>>>(SC, Lk, TH);
        gemm(SC, VT16, nullptr, AO, Lq, D, Lk, 1.f, 0);
        transp(P(3), WT16, D, D); gemm(AO, WT16, P(2), PR, Lq, D, D, 1.f, 0);
        add_inplace_kernel<<<blocks((size_t)Lq * D), 256, 0, stream>>>(X, PR, (size_t)Lq * D);

        // ---- FFN ----
        ln(X, P(17), P(16), T2, Lq);
        transp(P(9),  WT16, D, FFD); gemm(T2, WT16, P(8),  H,  Lq, FFD, D,   1.f, 1);
        transp(P(11), WT16, FFD, D); gemm(H,  WT16, P(10), PR, Lq, D,   FFD, 1.f, 0);
        add_inplace_kernel<<<blocks((size_t)Lq * D), 256, 0, stream>>>(X, PR, (size_t)Lq * D);
    }

    // Final LN straight into d_out, plus topk_percent scalar
    ln(X, fng, fnb, (float*)d_out, Lq);
    percent_kernel<<<1, 1, 0, stream>>>(topk, Lk, (float*)d_out + (size_t)Lq * D);
}
